// MoELayer_21036749816511
// MI455X (gfx1250) — matrix operand order repaired and checked by reading, not compiled, not run on hardware
//
#include <hip/hip_runtime.h>
#include <cstddef>

typedef __attribute__((ext_vector_type(16))) __bf16 v16bf;
typedef __attribute__((ext_vector_type(4)))  __bf16 v4bf;
typedef __attribute__((ext_vector_type(8)))  float  v8f;

#define Dm   512
#define FFm  2048
#define Em   16

// ---------------- activation ----------------
template<int ACT> __device__ __forceinline__ float actf(float v) {
  if constexpr (ACT == 1) return 0.5f * v * (1.f + erff(v * 0.7071067811865475f)); // exact GELU
  else if constexpr (ACT == 2) return v / (1.f + expf(-v));                        // SiLU
  else if constexpr (ACT == 3) return fmaxf(v, 0.f);                               // ReLU
  else if constexpr (ACT == 4) return 1.f / (1.f + expf(-v));                      // Sigmoid
  else return v;
}

// ---------------- zero out + counters ----------------
__global__ void zero_kernel(float* __restrict__ out, int n, int* __restrict__ counts) {
  int i = blockIdx.x * blockDim.x + threadIdx.x;
  if (i < n) out[i] = 0.f;
  if (i < Em) counts[i] = 0;
}

// ---------------- weight pre-pack: f32 [q][K][N] -> bf16 hi/lo in WMMA B-fragment order ----
// Fragment order: dst = q*K*N + ((nb*(K/32) + kc)*32 + lane)*16 + i
// where lane = (kk/16)*16 + (n%16), i = kk%16, kk = k%32, kc = k/32, nb = n/16.
__global__ void pack_w_kernel(const float* __restrict__ W, __bf16* __restrict__ Phi,
                              __bf16* __restrict__ Plo, int Kdim, int Ndim, int nExp) {
  size_t idx = (size_t)blockIdx.x * blockDim.x + threadIdx.x;
  size_t per = (size_t)Kdim * Ndim;
  size_t total = per * (size_t)nExp;
  if (idx >= total) return;
  int q = (int)(idx / per);
  size_t r = idx - (size_t)q * per;
  int k = (int)(r / Ndim), n = (int)(r - (size_t)(r / Ndim) * Ndim);
  float v = W[idx];
  __bf16 hi = (__bf16)v;
  __bf16 lo = (__bf16)(v - (float)hi);
  int nb = n >> 4, nl = n & 15, kc = k >> 5, kk = k & 31;
  int lane = ((kk >> 4) << 4) + nl;
  int i = kk & 15;
  size_t dst = (size_t)q * per +
               ((((size_t)nb * (Kdim >> 5) + kc) * 32 + lane) << 4) + i;
  Phi[dst] = hi;
  Plo[dst] = lo;
}

// ---------------- router: Linear -> LN -> GELU -> Linear -> softmax -> top2 ----------------
__global__ __launch_bounds__(256) void router_kernel(
    const float* __restrict__ x, const float* __restrict__ Wr1, const float* __restrict__ br1,
    const float* __restrict__ ln_g, const float* __restrict__ ln_b,
    const float* __restrict__ Wr2, const float* __restrict__ br2,
    const float* __restrict__ temp, const float* __restrict__ rbias, const float* __restrict__ gate,
    int* __restrict__ counts, int* __restrict__ stg_tok, float* __restrict__ stg_p, int T)
{
  __shared__ float xs[Dm];
  __shared__ float hs[Dm / 2];
  __shared__ float red[Dm / 2];
  __shared__ float lg[Em];
  const int t = blockIdx.x, tid = threadIdx.x;
  const float* xr = x + (size_t)t * Dm;
  xs[tid] = xr[tid];
  xs[tid + 256] = xr[tid + 256];
  __syncthreads();

  float acc = br1[tid];
  for (int d = 0; d < Dm; ++d) acc += xs[d] * Wr1[d * (Dm / 2) + tid];

  red[tid] = acc; __syncthreads();
  for (int s = 128; s > 0; s >>= 1) { if (tid < s) red[tid] += red[tid + s]; __syncthreads(); }
  float mu = red[0] * (1.f / 256.f); __syncthreads();
  float dv = acc - mu; red[tid] = dv * dv; __syncthreads();
  for (int s = 128; s > 0; s >>= 1) { if (tid < s) red[tid] += red[tid + s]; __syncthreads(); }
  float var = red[0] * (1.f / 256.f);
  float hn = (acc - mu) * (1.f / sqrtf(var + 1e-5f)) * ln_g[tid] + ln_b[tid];
  hn = 0.5f * hn * (1.f + erff(hn * 0.7071067811865475f));
  hs[tid] = hn; __syncthreads();

  if (tid < Em) {
    float l = br2[tid];
    for (int j = 0; j < Dm / 2; ++j) l += hs[j] * Wr2[j * Em + tid];
    lg[tid] = (l / temp[0] + rbias[tid]) * gate[tid];
  }
  __syncthreads();

  if (tid == 0) {
    float mx = lg[0];
    for (int e = 1; e < Em; ++e) mx = fmaxf(mx, lg[e]);
    float s = 0.f, p[Em];
    for (int e = 0; e < Em; ++e) { p[e] = expf(lg[e] - mx); s += p[e]; }
    float inv = 1.f / s;
    int e0 = 0, e1 = -1; float p0 = -1.f, p1 = -1.f;
    for (int e = 0; e < Em; ++e) {
      float v = p[e] * inv;
      if (v > p0)      { p1 = p0; e1 = e0; p0 = v; e0 = e; }
      else if (v > p1) { p1 = v;  e1 = e; }
    }
    int s0 = atomicAdd(&counts[e0], 1);
    stg_tok[e0 * T + s0] = t; stg_p[e0 * T + s0] = p0;
    int s1 = atomicAdd(&counts[e1], 1);
    stg_tok[e1 * T + s1] = t; stg_p[e1 * T + s1] = p1;
  }
}

// ---------------- segment offsets (pad each expert to 32 rows) ----------------
__global__ void segment_kernel(const int* __restrict__ counts, int* __restrict__ seg_off,
                               int* __restrict__ tile_expert, int nTiles) {
  if (blockIdx.x == 0 && threadIdx.x == 0) {
    for (int i = 0; i < nTiles; ++i) tile_expert[i] = -1;
    int acc = 0;
    for (int e = 0; e < Em; ++e) {
      seg_off[e] = acc;
      int pad = (counts[e] + 31) & ~31;
      for (int tl = acc >> 5; tl < (acc + pad) >> 5; ++tl) tile_expert[tl] = e;
      acc += pad;
    }
  }
}

// ---------------- compaction ----------------
__global__ void fill_kernel(const int* __restrict__ counts, const int* __restrict__ seg_off,
                            const int* __restrict__ stg_tok, const float* __restrict__ stg_p,
                            int* __restrict__ tok_idx, float* __restrict__ tok_p, int T) {
  int e = blockIdx.x;
  int cnt = counts[e], pad = (cnt + 31) & ~31, off = seg_off[e];
  for (int s = threadIdx.x; s < pad; s += blockDim.x) {
    tok_idx[off + s] = (s < cnt) ? stg_tok[e * T + s] : 0;
    tok_p  [off + s] = (s < cnt) ? stg_p  [e * T + s] : 0.f;
  }
}

// ---------------- bf16-split WMMA GEMM: 32 rows x 128 cols per block ----------------
// Ping-pong LDS A-fragment buffers + one-chunk-ahead register pipelining of the
// A global load; single barrier per K-chunk. B comes pre-packed from global in
// B-fragment order (contiguous 32B per lane per chunk).
template<int ACT, bool GATHER_X, bool SCATTER>
__global__ __launch_bounds__(256) void ffn_gemm(
    const float* __restrict__ Asrc, int ldA,
    const float* __restrict__ xsrc,
    const int* __restrict__ tok_idx, const float* __restrict__ tok_p,
    const int* __restrict__ tile_expert,
    const __bf16* __restrict__ Whi, const __bf16* __restrict__ Wlo,
    const float* __restrict__ btype,
    int Kdim, int Ndim, int typeSel,
    float* __restrict__ Hout, int ldH,
    float* __restrict__ out)
{
  const int bx = blockIdx.x;
  const int e = tile_expert[bx];
  if (e < 0 || (e % 3) != typeSel) return;
  const int q = e / 3;
  const size_t per = (size_t)Kdim * Ndim;
  const int nChunks = Kdim >> 5;

  const int tid = threadIdx.x;
  const int wave = tid >> 5, lane = tid & 31;
  const int nb = blockIdx.y * 8 + wave;          // 16-column block index
  const float* __restrict__ bias = btype + (size_t)q * Ndim;

  // packed B fragment stream for this wave: 512 bf16 (=1KB) per k-chunk, linear in kc
  const __bf16* __restrict__ bh =
      Whi + (size_t)q * per + ((((size_t)nb * nChunks) * 32 + lane) << 4);
  const __bf16* __restrict__ bl =
      Wlo + (size_t)q * per + ((((size_t)nb * nChunks) * 32 + lane) << 4);

  __shared__ __align__(32) __bf16 AH[2][2][32][16];   // [pingpong][subtile][lane][elem]
  __shared__ __align__(32) __bf16 AL[2][2][32][16];
  __shared__ int ArowIdx[32];
  if (tid < 32) {
    int slot = bx * 32 + tid;
    ArowIdx[tid] = GATHER_X ? tok_idx[slot] : slot;
  }
  __syncthreads();

  // Per-thread staging geometry: one float4 = 4 consecutive K of one row.
  const float* __restrict__ Abase = GATHER_X ? xsrc : Asrc;
  const int lda = GATHER_X ? Dm : ldA;
  const int mrow = tid >> 3;            // 0..31
  const int kq   = (tid & 7) << 2;      // 0,4,...,28
  const int rowoff = ArowIdx[mrow] * lda + kq;   // + kc*32 each chunk
  const int sub = mrow >> 4;
  const int ln  = ((kq >> 4) << 4) + (mrow & 15);
  const int ii  = kq & 15;

  v8f c0 = {}; v8f c1 = {};
  float4 va = *(const float4*)(Abase + rowoff);   // chunk 0, in flight

  for (int kc = 0; kc < nChunks; ++kc) {
    const int p = kc & 1;
    {   // convert the pre-loaded chunk and publish into ping-pong buffer p
      __bf16 h0 = (__bf16)va.x, h1 = (__bf16)va.y, h2 = (__bf16)va.z, h3 = (__bf16)va.w;
      v4bf hv = {h0, h1, h2, h3};
      v4bf lv = {(__bf16)(va.x - (float)h0), (__bf16)(va.y - (float)h1),
                 (__bf16)(va.z - (float)h2), (__bf16)(va.w - (float)h3)};
      *(v4bf*)(&AH[p][sub][ln][ii]) = hv;
      *(v4bf*)(&AL[p][sub][ln][ii]) = lv;
    }
    __syncthreads();
    if (kc + 1 < nChunks)   // next chunk's A load: latency hidden behind the WMMAs below
      va = *(const float4*)(Abase + rowoff + ((kc + 1) << 5));

    v16bf aH0 = *(const v16bf*)(&AH[p][0][lane][0]);
    v16bf aH1 = *(const v16bf*)(&AH[p][1][lane][0]);
    v16bf aL0 = *(const v16bf*)(&AL[p][0][lane][0]);
    v16bf aL1 = *(const v16bf*)(&AL[p][1][lane][0]);
    v16bf bHf = *(const v16bf*)(bh + ((size_t)kc << 9));
    v16bf bLf = *(const v16bf*)(bl + ((size_t)kc << 9));
    if (kc + 1 < nChunks) {   // hide next chunk's weight stream (global_prefetch_b8)
      __builtin_prefetch(bh + ((size_t)(kc + 1) << 9), 0, 1);
      __builtin_prefetch(bl + ((size_t)(kc + 1) << 9), 0, 1);
    }

    // split-bf16 fp32 emulation: AB ~= AhBh + AhBl + AlBh
    c0 = __builtin_amdgcn_wmma_f32_16x16x32_bf16(false, aH0, false, bHf, (short)0, c0, false, false);
    c0 = __builtin_amdgcn_wmma_f32_16x16x32_bf16(false, aH0, false, bLf, (short)0, c0, false, false);
    c0 = __builtin_amdgcn_wmma_f32_16x16x32_bf16(false, aL0, false, bHf, (short)0, c0, false, false);
    c1 = __builtin_amdgcn_wmma_f32_16x16x32_bf16(false, aH1, false, bHf, (short)0, c1, false, false);
    c1 = __builtin_amdgcn_wmma_f32_16x16x32_bf16(false, aH1, false, bLf, (short)0, c1, false, false);
    c1 = __builtin_amdgcn_wmma_f32_16x16x32_bf16(false, aL1, false, bHf, (short)0, c1, false, false);
  }

  const int n = (nb << 4) + (lane & 15);
  const int hh = lane >> 4;
  const float bv = bias[n];
#pragma unroll
  for (int r = 0; r < 8; ++r) {
    int row0 = hh * 8 + r;                 // C/D layout: lanes 0-15 M=r, lanes 16-31 M=8+r
    float v0 = actf<ACT>(c0[r] + bv);
    float v1 = actf<ACT>(c1[r] + bv);
    if constexpr (SCATTER) {
      int s0 = bx * 32 + row0, s1 = s0 + 16;
      atomicAdd(out + (size_t)tok_idx[s0] * Dm + n, tok_p[s0] * v0);
      atomicAdd(out + (size_t)tok_idx[s1] * Dm + n, tok_p[s1] * v1);
    } else {
      Hout[(size_t)(bx * 32 + row0) * ldH + n] = v0;
      Hout[(size_t)(bx * 32 + row0 + 16) * ldH + n] = v1;
    }
  }
}

// ---------------- host launch ----------------
extern "C" void kernel_launch(void* const* d_in, const int* in_sizes, int n_in,
                              void* d_out, int out_size, void* d_ws, size_t ws_size,
                              hipStream_t stream) {
  const float* x    = (const float*)d_in[0];
  const float* Wr1  = (const float*)d_in[1];
  const float* br1  = (const float*)d_in[2];
  const float* ln_g = (const float*)d_in[3];
  const float* ln_b = (const float*)d_in[4];
  const float* Wr2  = (const float*)d_in[5];
  const float* br2  = (const float*)d_in[6];
  const float* temp = (const float*)d_in[7];
  const float* rbias= (const float*)d_in[8];
  const float* gate = (const float*)d_in[9];
  const float* W0a = (const float*)d_in[10]; const float* b0a = (const float*)d_in[11];
  const float* W0b = (const float*)d_in[12]; const float* b0b = (const float*)d_in[13];
  const float* W1a = (const float*)d_in[14]; const float* b1a = (const float*)d_in[15];
  const float* W1b = (const float*)d_in[16]; const float* b1b = (const float*)d_in[17];
  const float* W1c = (const float*)d_in[18]; const float* b1c = (const float*)d_in[19];
  const float* W2a = (const float*)d_in[20]; const float* b2a = (const float*)d_in[21];
  const float* W2b = (const float*)d_in[22]; const float* b2b = (const float*)d_in[23];
  const float* W2c = (const float*)d_in[24]; const float* b2c = (const float*)d_in[25];
  float* out = (float*)d_out;

  const int T = in_sizes[0] / Dm;           // 2048 tokens
  const int cap = T * 2 + Em * 32;          // padded assignment capacity (mult of 32)
  const int nTiles = cap / 32;

  char* w = (char*)d_ws;
  auto carve = [&](size_t bytes) { void* p = (void*)w; w += (bytes + 255) & ~(size_t)255; return p; };
  int*   counts      = (int*)  carve(Em * 4);
  int*   seg_off     = (int*)  carve(Em * 4);
  int*   tile_expert = (int*)  carve((size_t)nTiles * 4);
  int*   tok_idx     = (int*)  carve((size_t)cap * 4);
  float* tok_p       = (float*)carve((size_t)cap * 4);
  int*   stg_tok     = (int*)  carve((size_t)Em * T * 4);
  float* stg_p       = (float*)carve((size_t)Em * T * 4);
  float* H1          = (float*)carve((size_t)cap * FFm * 4);
  float* H2          = (float*)carve((size_t)cap * FFm * 4);

  // packed weights (bf16 hi/lo), fragment order; same total bytes as f32 source
  auto carve_bf = [&](size_t elems) { return (__bf16*)carve(elems * 2); };
  const size_t s0a = (size_t)6 * Dm * FFm, s0b = (size_t)6 * FFm * Dm;
  const size_t s1a = (size_t)5 * Dm * FFm, s1b = (size_t)5 * FFm * FFm, s1c = (size_t)5 * FFm * Dm;
  const size_t s2a = (size_t)5 * Dm * FFm, s2b = (size_t)5 * FFm * Dm, s2c = (size_t)5 * Dm * Dm;
  __bf16 *P0aH = carve_bf(s0a), *P0aL = carve_bf(s0a);
  __bf16 *P0bH = carve_bf(s0b), *P0bL = carve_bf(s0b);
  __bf16 *P1aH = carve_bf(s1a), *P1aL = carve_bf(s1a);
  __bf16 *P1bH = carve_bf(s1b), *P1bL = carve_bf(s1b);
  __bf16 *P1cH = carve_bf(s1c), *P1cL = carve_bf(s1c);
  __bf16 *P2aH = carve_bf(s2a), *P2aL = carve_bf(s2a);
  __bf16 *P2bH = carve_bf(s2b), *P2bL = carve_bf(s2b);
  __bf16 *P2cH = carve_bf(s2c), *P2cL = carve_bf(s2c);

  auto packs = [&](const float* W, __bf16* H, __bf16* L, int K, int N, int nE) {
    size_t total = (size_t)nE * K * N;
    pack_w_kernel<<<(unsigned)((total + 255) / 256), 256, 0, stream>>>(W, H, L, K, N, nE);
  };
  packs(W0a, P0aH, P0aL, Dm, FFm, 6);
  packs(W0b, P0bH, P0bL, FFm, Dm, 6);
  packs(W1a, P1aH, P1aL, Dm, FFm, 5);
  packs(W1b, P1bH, P1bL, FFm, FFm, 5);
  packs(W1c, P1cH, P1cL, FFm, Dm, 5);
  packs(W2a, P2aH, P2aL, Dm, FFm, 5);
  packs(W2b, P2bH, P2bL, FFm, Dm, 5);
  packs(W2c, P2cH, P2cL, Dm, Dm, 5);

  zero_kernel<<<(out_size + 255) / 256, 256, 0, stream>>>(out, out_size, counts);
  router_kernel<<<T, 256, 0, stream>>>(x, Wr1, br1, ln_g, ln_b, Wr2, br2, temp, rbias, gate,
                                       counts, stg_tok, stg_p, T);
  segment_kernel<<<1, 32, 0, stream>>>(counts, seg_off, tile_expert, nTiles);
  fill_kernel<<<Em, 256, 0, stream>>>(counts, seg_off, stg_tok, stg_p, tok_idx, tok_p, T);

  dim3 blk(256, 1, 1);
  dim3 gFF(nTiles, FFm / 128, 1);
  dim3 gD(nTiles, Dm / 128, 1);

  // Layer 1 (gather x): act(x @ Wa + ba) -> H1
  ffn_gemm<1, true, false><<<gFF, blk, 0, stream>>>(nullptr, 0, x, tok_idx, tok_p, tile_expert,
                                                    P0aH, P0aL, b0a, Dm, FFm, 0, H1, FFm, nullptr);
  ffn_gemm<2, true, false><<<gFF, blk, 0, stream>>>(nullptr, 0, x, tok_idx, tok_p, tile_expert,
                                                    P1aH, P1aL, b1a, Dm, FFm, 1, H1, FFm, nullptr);
  ffn_gemm<3, true, false><<<gFF, blk, 0, stream>>>(nullptr, 0, x, tok_idx, tok_p, tile_expert,
                                                    P2aH, P2aL, b2a, Dm, FFm, 2, H1, FFm, nullptr);
  // Type 0 layer 2: H1 @ W0b + b0b -> scatter to out
  ffn_gemm<0, false, true><<<gD, blk, 0, stream>>>(H1, FFm, nullptr, tok_idx, tok_p, tile_expert,
                                                   P0bH, P0bL, b0b, FFm, Dm, 0, nullptr, 0, out);
  // Type 1 layer 2: H1 @ W1b + b1b -> H2 (no activation)
  ffn_gemm<0, false, false><<<gFF, blk, 0, stream>>>(H1, FFm, nullptr, tok_idx, tok_p, tile_expert,
                                                     P1bH, P1bL, b1b, FFm, FFm, 1, H2, FFm, nullptr);
  // Type 2 layer 2: sigmoid(H1 @ W2b + b2b) -> H2[:, :512]
  ffn_gemm<4, false, false><<<gD, blk, 0, stream>>>(H1, FFm, nullptr, tok_idx, tok_p, tile_expert,
                                                    P2bH, P2bL, b2b, FFm, Dm, 2, H2, FFm, nullptr);
  // Type 1 layer 3: H2 @ W1c + b1c -> scatter
  ffn_gemm<0, false, true><<<gD, blk, 0, stream>>>(H2, FFm, nullptr, tok_idx, tok_p, tile_expert,
                                                   P1cH, P1cL, b1c, FFm, Dm, 1, nullptr, 0, out);
  // Type 2 layer 3: H2 @ W2c + b2c -> scatter (K = 512)
  ffn_gemm<0, false, true><<<gD, blk, 0, stream>>>(H2, FFm, nullptr, tok_idx, tok_p, tile_expert,
                                                   P2cH, P2cL, b2c, Dm, Dm, 2, nullptr, 0, out);
}
